// ConvLSTM_77764677861656
// MI455X (gfx1250) — matrix operand order repaired and checked by reading, not compiled, not run on hardware
//
#include <hip/hip_runtime.h>
#include <hip/hip_bf16.h>
#include <stdint.h>

typedef __attribute__((ext_vector_type(8)))  _Float16 v8h;
typedef __attribute__((ext_vector_type(16))) _Float16 v16h;
typedef __attribute__((ext_vector_type(8)))  float    v8f;

union Frag16 { v16h v; v8h h[2]; };

__device__ __forceinline__ float hsig(float x) {
    float y = 0.2f * x + 0.5f;
    return fminf(fmaxf(y, 0.0f), 1.0f);
}

// ---------------------------------------------------------------- utilities
__global__ void k_zero(uint32_t* __restrict__ p, long n) {
    long i = (long)blockIdx.x * blockDim.x + threadIdx.x;
    if (i < n) p[i] = 0u;
}

// LayerNorm over last-dim=16, f32 -> f16  (one thread per pixel)
__global__ void k_ln16(const float* __restrict__ x, const float* __restrict__ g,
                       const float* __restrict__ b, _Float16* __restrict__ o, long npix) {
    long p = (long)blockIdx.x * blockDim.x + threadIdx.x;
    if (p >= npix) return;
    const float* xp = x + p * 16;
    float mu = 0.f;
#pragma unroll
    for (int i = 0; i < 16; ++i) mu += xp[i];
    mu *= (1.f / 16.f);
    float var = 0.f;
#pragma unroll
    for (int i = 0; i < 16; ++i) { float d = xp[i] - mu; var += d * d; }
    var *= (1.f / 16.f);
    float rs = rsqrtf(var + 1e-3f);
#pragma unroll
    for (int i = 0; i < 16; ++i)
        o[p * 16 + i] = (_Float16)((xp[i] - mu) * rs * g[i] + b[i]);
}

// Repack f32 weights [K,256] row-major into WMMA B-fragment order (f16),
// layout [nTile][kTile][lane][16 halves]; lanes 0-15 hold K 0..7,16..23 of the
// k-tile (n = lane), lanes 16-31 hold K 8..15,24..31 (ISA 7.12.2 table order).
__global__ void k_repack(const float* __restrict__ w, _Float16* __restrict__ out,
                         int K, int kTiles) {
    long idx = (long)blockIdx.x * blockDim.x + threadIdx.x;
    long total = (long)16 * kTiles * 512;
    if (idx >= total) return;
    int e    = (int)(idx & 15);
    int lane = (int)((idx >> 4) & 31);
    long t2  = idx >> 9;
    int kt   = (int)(t2 % kTiles);
    int nt   = (int)(t2 / kTiles);
    int n = nt * 16 + (lane & 15);
    int k = kt * 32 + (e >> 3) * 16 + (lane >> 4) * 8 + (e & 7);
    float v = (k < K) ? w[(long)k * 256 + n] : 0.f;
    out[idx] = (_Float16)v;
}

// ------------------------------------------------ CIN=16 implicit-GEMM conv
// VALID 5x5 conv, K=400 (13 k-tiles, patch 25 zero). One wave per 16x16 tile,
// 8 waves/block = 128 rows of M; gridDim.y = 16 n-tiles.
template<int WIN, int HOUT, int WOUT, int KTILES>
__global__ __launch_bounds__(256)
void k_conv_gemm16(const _Float16* __restrict__ src, long batch_stride,
                   const _Float16* __restrict__ wp,
                   const float* __restrict__ bias, float* __restrict__ g) {
    __shared__ _Float16 bpanel[KTILES * 512];
    const int tid  = threadIdx.x;
    const int lane = tid & 31;
    const int wave = tid >> 5;
    const int nTile = blockIdx.y;
    {
        const v8h* s = (const v8h*)(wp + (long)nTile * KTILES * 512);
        v8h* d = (v8h*)bpanel;
#pragma unroll 4
        for (int i = tid; i < KTILES * 64; i += 256) d[i] = s[i];
    }
    __syncthreads();

    const int mTileBase = (blockIdx.x * 8 + wave) * 16;
    const int n  = nTile * 16 + (lane & 15);
    const int mA = mTileBase + (lane & 15);
    const int ox = mA % WOUT;
    const int t0 = mA / WOUT;
    const int oy = t0 % HOUT;
    const int bb = t0 / HOUT;
    const _Float16* pix0 = src + (long)bb * batch_stride + ((long)oy * WIN + ox) * 16;
    const int hi = (lane >> 4) * 8;
    const _Float16* bp = bpanel + lane * 16;

    v8f acc = {};
#pragma unroll
    for (int kt = 0; kt < KTILES; ++kt) {
        const int p0 = 2 * kt, p1 = 2 * kt + 1;
        Frag16 a; v8h z = {};
        {
            const int ky = p0 / 5, kx = p0 % 5;
            a.h[0] = *(const v8h*)(pix0 + (ky * WIN + kx) * 16 + hi);
        }
        if (p1 < 25) {
            const int ky = p1 / 5, kx = p1 % 5;
            a.h[1] = *(const v8h*)(pix0 + (ky * WIN + kx) * 16 + hi);
        } else {
            a.h[1] = z;
        }
        Frag16 bf;
        bf.h[0] = *(const v8h*)(bp + kt * 512);
        bf.h[1] = *(const v8h*)(bp + kt * 512 + 8);
        acc = __builtin_amdgcn_wmma_f32_16x16x32_f16(
                  false, a.v, false, bf.v, (short)0, acc, false, false);
    }

    const int mD = mTileBase + 8 * (lane >> 4);
    float* gp = g + (long)mD * 256 + n;
    float bv = bias[n];
#pragma unroll
    for (int r = 0; r < 8; ++r) gp[(long)r * 256] = acc[r] + bv;
}

// ------------------------------------------------ CIN=64 implicit-GEMM conv
// M2xN4-blocked: each wave computes a 32x64 output tile (8 WMMA accumulators).
// Per WMMA: 0.5 global_load_b128 (A reused over 4 n-tiles) and
//           1.0 ds_load_b128    (B reused over 2 m-tiles).
// B staged per-ky chunk: 4 n-tiles x 10 k-tiles = 40KB LDS.
// Block = 128 threads (4 waves) covering 128 rows of M; gridDim.y = 4 n-groups.
template<int HIN, int WIN, int HOUT, int WOUT, int PAD, int KTILES>
__global__ __launch_bounds__(128)
void k_conv_gemm64(const _Float16* __restrict__ src, long batch_stride,
                   const _Float16* __restrict__ wp,
                   const float* __restrict__ bias, float* __restrict__ g,
                   int accumulate) {
    __shared__ _Float16 bchunk[4 * 10 * 512];      // 40 KB
    const int tid  = threadIdx.x;
    const int lane = tid & 31;
    const int wave = tid >> 5;
    const int ng   = blockIdx.y;                   // n-group of 4 n-tiles

    const int mTileBase = (blockIdx.x * 4 + wave) * 32;   // 2 m-tiles per wave

    int oxv[2], oyv[2];
    const _Float16* pix0v[2];
#pragma unroll
    for (int i = 0; i < 2; ++i) {
        const int mA = mTileBase + i * 16 + (lane & 15);
        const int ox = mA % WOUT;
        const int t0 = mA / WOUT;
        const int oy = t0 % HOUT;
        const int bb = t0 / HOUT;
        oxv[i] = ox; oyv[i] = oy;
        pix0v[i] = src + (long)bb * batch_stride + ((long)oy * WIN + ox) * 64;
    }
    const int hi = (lane >> 4) * 8;

    v8f zf = {};
    v8f acc[2][4] = {{zf, zf, zf, zf}, {zf, zf, zf, zf}};
    const v8h* wsrc = (const v8h*)wp;

#pragma unroll
    for (int ky = 0; ky < 5; ++ky) {
        __syncthreads();
        {   // stage B: 4 n-tiles x k-tiles [ky*10, ky*10+10)
            v8h* d = (v8h*)bchunk;
#pragma unroll
            for (int i = tid; i < 2560; i += 128) {
                const int j = i / 640, r = i - j * 640;
                d[i] = wsrc[((long)(4 * ng + j) * KTILES + ky * 10) * 64 + r];
            }
        }
        __syncthreads();
#pragma unroll
        for (int kx = 0; kx < 5; ++kx) {
            const int dy = ky - PAD, dx = kx - PAD;
            bool validv[2];
            const _Float16* pv[2];
#pragma unroll
            for (int i = 0; i < 2; ++i) {
                pv[i] = pix0v[i] + (dy * WIN + dx) * 64;
                validv[i] = true;
                if (PAD != 0)
                    validv[i] = ((unsigned)(oyv[i] + dy) < (unsigned)HIN) &&
                                ((unsigned)(oxv[i] + dx) < (unsigned)WIN);
            }
#pragma unroll
            for (int half = 0; half < 2; ++half) {
                Frag16 a[2];
#pragma unroll
                for (int i = 0; i < 2; ++i) {
                    v8h z = {};
                    a[i].h[0] = z; a[i].h[1] = z;
                    if (validv[i]) {
                        a[i].h[0] = *(const v8h*)(pv[i] + half * 32 + hi);
                        a[i].h[1] = *(const v8h*)(pv[i] + half * 32 + 16 + hi);
                    }
                }
                const int lkt = kx * 2 + half;
#pragma unroll
                for (int j = 0; j < 4; ++j) {
                    Frag16 bf;
                    const _Float16* bq = bchunk + (j * 10 + lkt) * 512 + lane * 16;
                    bf.h[0] = *(const v8h*)bq;
                    bf.h[1] = *(const v8h*)(bq + 8);
                    acc[0][j] = __builtin_amdgcn_wmma_f32_16x16x32_f16(
                                    false, a[0].v, false, bf.v, (short)0, acc[0][j], false, false);
                    acc[1][j] = __builtin_amdgcn_wmma_f32_16x16x32_f16(
                                    false, a[1].v, false, bf.v, (short)0, acc[1][j], false, false);
                }
            }
        }
    }

#pragma unroll
    for (int i = 0; i < 2; ++i) {
        const int mD = mTileBase + i * 16 + 8 * (lane >> 4);
#pragma unroll
        for (int j = 0; j < 4; ++j) {
            const int n = (4 * ng + j) * 16 + (lane & 15);
            float* gp = g + (long)mD * 256 + n;
            if (accumulate) {
#pragma unroll
                for (int r = 0; r < 8; ++r) gp[(long)r * 256] += acc[i][j][r];
            } else {
                float bv = bias[n];
#pragma unroll
                for (int r = 0; r < 8; ++r) gp[(long)r * 256] = acc[i][j][r] + bv;
            }
        }
    }
}

// --------------------------------------- layer-1 gates + fused LayerNorm(64)
// One wave32 per pixel; each lane owns channels {lane, lane+32}.
__global__ void k_gates1(const float* __restrict__ g, float* __restrict__ c,
                         _Float16* __restrict__ h16, _Float16* __restrict__ lnh,
                         const float* __restrict__ ln_g, const float* __restrict__ ln_b,
                         int npix) {
    int w = (int)(((long)blockIdx.x * blockDim.x + threadIdx.x) >> 5);
    int lane = threadIdx.x & 31;
    if (w >= npix) return;
    const float* gp = g + (long)w * 256;
    float hv[2];
#pragma unroll
    for (int j = 0; j < 2; ++j) {
        int f = lane + j * 32;
        float i_ = hsig(gp[f]);
        float f_ = hsig(gp[64 + f]);
        float cg = tanhf(gp[128 + f]);
        float o_ = hsig(gp[192 + f]);
        long ci = (long)w * 64 + f;
        float cv = f_ * c[ci] + i_ * cg;
        c[ci] = cv;
        float h = o_ * tanhf(cv);
        hv[j] = h;
        h16[ci] = (_Float16)h;
    }
    float s  = hv[0] + hv[1];
    float sq = hv[0] * hv[0] + hv[1] * hv[1];
#pragma unroll
    for (int off = 16; off >= 1; off >>= 1) {
        s  += __shfl_xor(s,  off, 32);
        sq += __shfl_xor(sq, off, 32);
    }
    float mu  = s * (1.f / 64.f);
    float var = sq * (1.f / 64.f) - mu * mu;
    float rs  = rsqrtf(var + 1e-3f);
#pragma unroll
    for (int j = 0; j < 2; ++j) {
        int f = lane + j * 32;
        lnh[(long)w * 64 + f] = (_Float16)((hv[j] - mu) * rs * ln_g[f] + ln_b[f]);
    }
}

// ------------------------------------------------------ layer-2 gates (no LN)
__global__ void k_gates2(const float* __restrict__ g, float* __restrict__ c,
                         _Float16* __restrict__ h16, float* __restrict__ h32,
                         long total) {
    long idx = (long)blockIdx.x * blockDim.x + threadIdx.x;
    if (idx >= total) return;
    int f = (int)(idx & 63);
    long p = idx >> 6;
    const float* gp = g + p * 256;
    float i_ = hsig(gp[f]);
    float f_ = hsig(gp[64 + f]);
    float cg = tanhf(gp[128 + f]);
    float o_ = hsig(gp[192 + f]);
    float cv = f_ * c[idx] + i_ * cg;
    c[idx] = cv;
    float h = o_ * tanhf(cv);
    h16[idx] = (_Float16)h;
    h32[idx] = h;
}

// ------------------------------------------------- mean-pool + dense + scale
__global__ void k_pool_dense(const float* __restrict__ h, const float* __restrict__ dw,
                             const float* __restrict__ db, float* __restrict__ out) {
    int b = blockIdx.x, tid = threadIdx.x;
    float s = 0.f;
    for (long i = tid; i < 1024 * 64; i += 256) {
        int f = (int)(i & 63);
        s += h[(long)b * 65536 + i] * dw[f];
    }
    __shared__ float red[256];
    red[tid] = s;
    __syncthreads();
    for (int st = 128; st >= 1; st >>= 1) {
        if (tid < st) red[tid] += red[tid + st];
        __syncthreads();
    }
    if (tid == 0) out[b] = (red[0] * (1.f / 1024.f) + db[0]) * 0.1f;
}

// ===========================================================================
extern "C" void kernel_launch(void* const* d_in, const int* in_sizes, int n_in,
                              void* d_out, int out_size, void* d_ws, size_t ws_size,
                              hipStream_t stream) {
    (void)in_sizes; (void)n_in; (void)out_size; (void)ws_size;
    const float* x     = (const float*)d_in[0];
    const float* ln_g1 = (const float*)d_in[1];
    const float* ln_b1 = (const float*)d_in[2];
    const float* W1    = (const float*)d_in[3];
    const float* U1    = (const float*)d_in[4];
    const float* b1    = (const float*)d_in[5];
    const float* ln_g2 = (const float*)d_in[6];
    const float* ln_b2 = (const float*)d_in[7];
    const float* W2    = (const float*)d_in[8];
    const float* U2    = (const float*)d_in[9];
    const float* b2    = (const float*)d_in[10];
    const float* dw    = (const float*)d_in[11];
    const float* db    = (const float*)d_in[12];
    float* out = (float*)d_out;

    // ---- workspace layout (bytes, 256-aligned cursor) ----
    char* ws = (char*)d_ws;
    size_t cur = 0;
    auto alloc = [&](size_t bytes) { void* p = ws + cur;
        cur += (bytes + 255) & ~(size_t)255; return p; };

    const long LNX   = 3276800;           // 8*16*40*40*16 halves
    const int  KT1   = 13;                // K=400 -> pad 416
    const int  KTR   = 50;                // K=1600
    const long NPIX1 = 10368;             // 8*36*36
    const long NPIX2 = 8192;              // 8*32*32

    _Float16* lnx   = (_Float16*)alloc(LNX * 2);
    _Float16* w1p   = (_Float16*)alloc((size_t)16 * KT1 * 512 * 2);
    _Float16* u1p   = (_Float16*)alloc((size_t)16 * KTR * 512 * 2);
    _Float16* w2p   = (_Float16*)alloc((size_t)16 * KTR * 512 * 2);
    _Float16* u2p   = (_Float16*)alloc((size_t)16 * KTR * 512 * 2);
    float*    g1    = (float*)alloc((size_t)NPIX1 * 256 * 4);
    _Float16* h1f   = (_Float16*)alloc((size_t)NPIX1 * 64 * 2);
    float*    c1    = (float*)alloc((size_t)NPIX1 * 64 * 4);
    _Float16* lnhs  = (_Float16*)alloc((size_t)16 * NPIX1 * 64 * 2);  // [T][B,36,36,64]
    float*    g2    = (float*)alloc((size_t)NPIX2 * 256 * 4);
    _Float16* h2f   = (_Float16*)alloc((size_t)NPIX2 * 64 * 2);
    float*    c2    = (float*)alloc((size_t)NPIX2 * 64 * 4);
    float*    h2o   = (float*)alloc((size_t)NPIX2 * 64 * 4);

    // ---- LayerNorm 1 (f32 -> f16) ----
    k_ln16<<<dim3(800), dim3(256), 0, stream>>>(x, ln_g1, ln_b1, lnx, 204800);

    // ---- weight repacks into WMMA B-fragment order ----
    {
        long t1 = (long)16 * KT1 * 512, tr = (long)16 * KTR * 512;
        k_repack<<<dim3((unsigned)((t1 + 255) / 256)), dim3(256), 0, stream>>>(W1, w1p, 400, KT1);
        k_repack<<<dim3((unsigned)((tr + 255) / 256)), dim3(256), 0, stream>>>(U1, u1p, 1600, KTR);
        k_repack<<<dim3((unsigned)((tr + 255) / 256)), dim3(256), 0, stream>>>(W2, w2p, 1600, KTR);
        k_repack<<<dim3((unsigned)((tr + 255) / 256)), dim3(256), 0, stream>>>(U2, u2p, 1600, KTR);
    }

    // ---- zero recurrent state (h0 = c0 = 0) ----
    auto zero = [&](void* p, long bytes) {
        long n = bytes / 4;
        k_zero<<<dim3((unsigned)((n + 255) / 256)), dim3(256), 0, stream>>>((uint32_t*)p, n);
    };
    zero(h1f, NPIX1 * 64 * 2);
    zero(c1,  NPIX1 * 64 * 4);
    zero(h2f, NPIX2 * 64 * 2);
    zero(c2,  NPIX2 * 64 * 4);

    // ---- ConvLSTM layer 1 : 40x40x16 -> (valid) 36x36x64, T steps ----
    for (int t = 0; t < 16; ++t) {
        // input conv (VALID, C=16, K=400->416)
        k_conv_gemm16<40, 36, 36, 13><<<dim3(81, 16), dim3(256), 0, stream>>>(
            lnx + (long)t * 25600, 409600, w1p, b1, g1);
        // recurrent conv (SAME, C=64, K=1600), accumulate
        k_conv_gemm64<36, 36, 36, 36, 2, 50><<<dim3(81, 4), dim3(128), 0, stream>>>(
            h1f, (long)36 * 36 * 64, u1p, nullptr, g1, 1);
        // gates + cell update + fused LayerNorm(64)
        k_gates1<<<dim3(1296), dim3(256), 0, stream>>>(
            g1, c1, h1f, lnhs + (long)t * NPIX1 * 64, ln_g2, ln_b2, (int)NPIX1);
    }

    // ---- ConvLSTM layer 2 : 36x36x64 -> (valid) 32x32x64, T steps ----
    for (int t = 0; t < 16; ++t) {
        k_conv_gemm64<36, 36, 32, 32, 0, 50><<<dim3(64, 4), dim3(128), 0, stream>>>(
            lnhs + (long)t * NPIX1 * 64, (long)36 * 36 * 64, w2p, b2, g2, 0);
        k_conv_gemm64<32, 32, 32, 32, 2, 50><<<dim3(64, 4), dim3(128), 0, stream>>>(
            h2f, (long)32 * 32 * 64, u2p, nullptr, g2, 1);
        k_gates2<<<dim3(2048), dim3(256), 0, stream>>>(
            g2, c2, h2f, h2o, NPIX2 * 64);
    }

    // ---- mean pool + dense + 0.1 scale ----
    k_pool_dense<<<dim3(8), dim3(256), 0, stream>>>(h2o, dw, db, out);
}
